// OriginalDiffMPO_33964601376806
// MI455X (gfx1250) — matrix operand order repaired and checked by reading, not compiled, not run on hardware
//
#include <hip/hip_runtime.h>
#include <hip/hip_bf16.h>

// ---------------------------------------------------------------------------
// MPO projected-gradient solver for MI455X (gfx1250).
//
// Phase 1 (WMMA): Sigma = L @ L^T per (b,h) -- 1536 independent 100x100x100
//     matmuls via v_wmma_f32_16x16x32_f16 (f32 accumulate). L staged in LDS as
//     zero-padded f16; fragments built with ds_load_b128; Sigma row-padded to
//     112 so all D stores are unpredicated (no EXEC churn).
// Phase 2 (solver): 500 projected-gradient iterations, one workgroup per
//     batch b, one wave32 per horizon step h; w vectors live in LDS, Sigma
//     streamed from L2 (88 MB < 192 MB L2); exact simplex projection via
//     bisection with wave32 shuffle butterflies.
// ---------------------------------------------------------------------------

typedef __attribute__((ext_vector_type(16))) _Float16 v16h;
typedef __attribute__((ext_vector_type(8)))  _Float16 v8h;
typedef __attribute__((ext_vector_type(8)))  float    v8f;

#define BB     128
#define HH     12
#define NN     100
#define SPAD   128            // padded Sigma row length (floats)
#define SROWS  112            // padded Sigma row count (rows 100..111 unused)
#define NTILE  7              // ceil(100/16)
#define N_ITERS 500

#define LPAD   112            // padded L rows/cols for WMMA tiles
#define LROW   136            // LDS row stride in halves (272 B -> conflict-free)

#define GAMMA      5.0f
#define COST_COEFF 1e-3f
#define KAPPA      2.0627128075074256f
#define PEN_XI     100.0f
#define LR         0.01f

// ---------------------------------------------------------------------------
// Kernel A: Sigma[bh] = L[bh] @ L[bh]^T  (f16 inputs, f32 accumulate, WMMA)
// grid = B*H blocks, 224 threads (7 waves); wave w owns tile-row ti = w.
// ---------------------------------------------------------------------------
__global__ __launch_bounds__(224) void sigma_llt_kernel(
    const float* __restrict__ L, float* __restrict__ S)
{
    __shared__ __align__(16) _Float16 Ll[LPAD * LROW];

    const int bh   = blockIdx.x;                  // 0..1535
    const int lane = threadIdx.x & 31;
    const int ti   = threadIdx.x >> 5;            // 0..6
    const float* __restrict__ Lb = L + (size_t)bh * (NN * NN);
    float* __restrict__       Sb = S + (size_t)bh * (SROWS * SPAD);

    // ---- zero-fill the padded f16 staging buffer, then stage-convert L ----
    for (int i = threadIdx.x; i < (LPAD * LROW) / 2; i += 224)
        ((unsigned int*)Ll)[i] = 0u;
    __syncthreads();
    for (int i = threadIdx.x; i < NN * NN; i += 224) {
        const int row = i / NN;
        const int c   = i - row * NN;
        Ll[row * LROW + c] = (_Float16)Lb[i];
    }
    __syncthreads();

    const int l15   = lane & 15;
    const int rowA  = ti * 16 + l15;
    const int base8 = (lane < 16) ? 0 : 8;        // A-layout half-wave K offset
    const int kbB   = (lane < 16) ? 0 : 8;        // B-layout half-wave K offset

    // ---- A fragments: k = kc*32 + base8 + {0..7} and +16+{0..7} (AMD table order)
    const _Float16* baseA = Ll + rowA * LROW;
    v16h afrag[4];
#pragma unroll
    for (int kc = 0; kc < 4; ++kc) {
        const v8h lo = *(const v8h*)(baseA + kc * 32 + base8);
        const v8h hi = *(const v8h*)(baseA + kc * 32 + base8 + 16);
        afrag[kc] = __builtin_shufflevector(lo, hi,
            0, 1, 2, 3, 4, 5, 6, 7, 8, 9, 10, 11, 12, 13, 14, 15);
    }

    for (int tj = 0; tj < NTILE; ++tj) {
        const int rowB = tj * 16 + l15;           // B holds L^T: col j <- row of L
        const _Float16* baseB = Ll + rowB * LROW + kbB;

        // Hoist all B-fragment loads ahead of the WMMA chain so the LDS
        // fetches overlap the serially-dependent accumulator WMMAs.
        v16h bfrag[4];
#pragma unroll
        for (int kc = 0; kc < 4; ++kc) {
            const v8h lo = *(const v8h*)(baseB + kc * 32);
            const v8h hi = *(const v8h*)(baseB + kc * 32 + 16);
            bfrag[kc] = __builtin_shufflevector(lo, hi,
                0, 1, 2, 3, 4, 5, 6, 7, 8, 9, 10, 11, 12, 13, 14, 15);
        }

        v8f acc = {};
#pragma unroll
        for (int kc = 0; kc < 4; ++kc)
            acc = __builtin_amdgcn_wmma_f32_16x16x32_f16(
                false, afrag[kc], false, bfrag[kc], (short)0, acc, false, false);

        // C/D layout: VGPR r -> M = r (lanes 0-15) / r+8 (lanes 16-31); N = lane&15.
        // Sigma has 112 padded rows -> unconditional, fully coalesced stores.
        const int col = tj * 16 + l15;            // cols 100..111 hold exact zeros
        const int rb  = ti * 16 + ((lane < 16) ? 0 : 8);
#pragma unroll
        for (int r = 0; r < 8; ++r)
            Sb[(rb + r) * SPAD + col] = acc[r];
    }

    // Zero the remaining pad columns 112..127 of rows 0..99 (ws is poisoned).
    for (int idx = threadIdx.x; idx < NN * 16; idx += 224) {
        const int row = idx >> 4;
        const int c   = 112 + (idx & 15);
        Sb[row * SPAD + c] = 0.0f;
    }
}

// ---------------------------------------------------------------------------
// Kernel B: projected-gradient solver. grid = B blocks, 384 threads (12 waves,
// wave h owns horizon step h). Lane owns columns n = 4*lane + s, s in 0..3.
// ---------------------------------------------------------------------------
__global__ __launch_bounds__(384) void mpo_solver_kernel(
    const float* __restrict__ mu, const float* __restrict__ w_prev,
    const float* __restrict__ cvar, const float* __restrict__ S,
    float* __restrict__ out)
{
    const int b    = blockIdx.x;
    const int h    = threadIdx.x >> 5;            // 0..11
    const int lane = threadIdx.x & 31;
    const int n0   = lane * 4;

    __shared__ __align__(16) float wsh[HH + 1][SPAD]; // [0]=w_prev, [1+h]=w_h

    const float* __restrict__ Srow = S + (size_t)(b * HH + h) * (SROWS * SPAD);
    const float limit = cvar[b];

    // Warm L2 with this wave's 50 KB Sigma slab (global_prefetch_b8 path).
    for (int i = lane; i < (NN * SPAD * 4) / 128; i += 32)
        __builtin_prefetch(((const char*)Srow) + (size_t)i * 128, 0, 2);

    float mur[4];
#pragma unroll
    for (int s = 0; s < 4; ++s) {
        const int n   = n0 + s;
        const bool ok = (n < NN);
        mur[s]        = ok ? mu[(b * HH + h) * NN + n] : 0.0f;
        const float wp = ok ? w_prev[b * NN + n] : 0.0f;
        wsh[h + 1][n] = wp;                       // pad slots init to 0
        if (h == 0) wsh[0][n] = wp;
    }
    __syncthreads();

    for (int iter = 0; iter < N_ITERS; ++iter) {
        // ---- q = Sigma * w  (coalesced b128 loads; pad cols are zero) ----
        float q0 = 0.f, q1 = 0.f, q2 = 0.f, q3 = 0.f;
#pragma unroll 5
        for (int n4 = 0; n4 < 25; ++n4) {
            const float4 wv = *(const float4*)&wsh[h + 1][n4 * 4];
            const float* r = Srow + (size_t)(n4 * 4) * SPAD + n0;
            const float4 s0 = *(const float4*)(r);
            const float4 s1 = *(const float4*)(r + SPAD);
            const float4 s2 = *(const float4*)(r + 2 * SPAD);
            const float4 s3 = *(const float4*)(r + 3 * SPAD);
            q0 += s0.x * wv.x + s1.x * wv.y + s2.x * wv.z + s3.x * wv.w;
            q1 += s0.y * wv.x + s1.y * wv.y + s2.y * wv.z + s3.y * wv.w;
            q2 += s0.z * wv.x + s1.z * wv.y + s2.z * wv.z + s3.z * wv.w;
            q3 += s0.w * wv.x + s1.w * wv.y + s2.w * wv.z + s3.w * wv.w;
        }
        float qv[4] = {q0, q1, q2, q3};

        // ---- ret = mu.w, quad = w.Sigma.w  (wave32 butterfly reduce) ----
        float pr = 0.f, pq = 0.f;
#pragma unroll
        for (int s = 0; s < 4; ++s) {
            const float w = wsh[h + 1][n0 + s];   // 0 in pad slots
            pr += mur[s] * w;
            pq += w * qv[s];                      // qv==0 in pad cols
        }
#pragma unroll
        for (int off = 16; off > 0; off >>= 1) {
            pr += __shfl_xor(pr, off, 32);
            pq += __shfl_xor(pq, off, 32);
        }
        const float sigma   = sqrtf(pq + 1e-12f);
        const float violraw = -pr + KAPPA * sigma - limit;
        const bool  act     = (violraw > 0.0f);
        const float cq = 2.0f * GAMMA + (act ? PEN_XI * KAPPA / sigma : 0.0f);
        const float cm = act ? (1.0f + PEN_XI) : 1.0f;

        // ---- gradient + GD candidate ----
        float v4[4];
#pragma unroll
        for (int s = 0; s < 4; ++s) {
            const int n = n0 + s;
            const float w  = wsh[h + 1][n];
            const float wm = wsh[h][n];           // lagged (h==0 -> w_prev)
            const float d1 = w - wm;
            const float r1 = d1 * rsqrtf(d1 * d1 + 1e-10f);
            float r2 = 0.0f;
            if (h < HH - 1) {
                const float d2 = wsh[h + 2][n] - w;
                r2 = d2 * rsqrtf(d2 * d2 + 1e-10f);
            }
            const float g = -cm * mur[s] + cq * qv[s] + COST_COEFF * (r1 - r2);
            v4[s] = (n < NN) ? (w - LR * g) : -1e30f;  // pad slots never activate
        }
        __syncthreads();   // all reads of old w complete before overwriting

        // ---- exact simplex projection via bisection on theta ----
        float pm = fmaxf(fmaxf(v4[0], v4[1]), fmaxf(v4[2], v4[3]));
#pragma unroll
        for (int off = 16; off > 0; off >>= 1)
            pm = fmaxf(pm, __shfl_xor(pm, off, 32));
        float lo = pm - 1.0f, hi = pm;            // f(lo)>=0 > f(hi)=-1
        for (int it = 0; it < 32; ++it) {
            const float th = 0.5f * (lo + hi);
            float ps = fmaxf(v4[0] - th, 0.f) + fmaxf(v4[1] - th, 0.f)
                     + fmaxf(v4[2] - th, 0.f) + fmaxf(v4[3] - th, 0.f);
#pragma unroll
            for (int off = 16; off > 0; off >>= 1)
                ps += __shfl_xor(ps, off, 32);
            if (ps >= 1.0f) lo = th; else hi = th;
        }
        const float theta = 0.5f * (lo + hi);
#pragma unroll
        for (int s = 0; s < 4; ++s)
            wsh[h + 1][n0 + s] = fmaxf(v4[s] - theta, 0.0f);
        __syncthreads();
    }

    // ---- emit w (B,H,N) ----
#pragma unroll
    for (int s = 0; s < 4; ++s) {
        const int n = n0 + s;
        if (n < NN) out[(size_t)(b * HH + h) * NN + n] = wsh[h + 1][n0 + s];
    }
}

// ---------------------------------------------------------------------------
extern "C" void kernel_launch(void* const* d_in, const int* in_sizes, int n_in,
                              void* d_out, int out_size, void* d_ws, size_t ws_size,
                              hipStream_t stream)
{
    const float* mu      = (const float*)d_in[0];  // (B,H,N)
    const float* L       = (const float*)d_in[1];  // (B,H,N,N)
    const float* w_prev  = (const float*)d_in[2];  // (B,N)
    const float* cvar    = (const float*)d_in[3];  // (B,)
    float*       out     = (float*)d_out;          // (B,H,N)
    float*       Sigma   = (float*)d_ws;           // B*H*SROWS*SPAD floats (~88 MB)

    sigma_llt_kernel<<<dim3(BB * HH), dim3(224), 0, stream>>>(L, Sigma);
    mpo_solver_kernel<<<dim3(BB), dim3(384), 0, stream>>>(mu, w_prev, cvar, Sigma, out);
}
